// DecoderLayer_17575006175316
// MI455X (gfx1250) — matrix operand order repaired and checked by reading, not compiled, not run on hardware
//
#include <hip/hip_runtime.h>
#include <cstddef>
#include <cstdint>

// ---------------------------------------------------------------------------
// CDNA5 (gfx1250) decoder layer: bf16 WMMA GEMMs + flash attention.
// ---------------------------------------------------------------------------

typedef __bf16 bf16;
typedef __attribute__((ext_vector_type(16))) __bf16 v16bf;
typedef __attribute__((ext_vector_type(8)))  __bf16 v8bf;
typedef __attribute__((ext_vector_type(8)))  float  v8f;

#define WMMA_BF16(a, b, c) \
  __builtin_amdgcn_wmma_f32_16x16x32_bf16(false, (a), false, (b), (short)0, (c), false, false)

static constexpr int D_MODEL  = 512;
static constexpr int N_HEADS  = 8;
static constexpr int HEAD_DIM = 64;
static constexpr int FF_DIM   = 2048;
static constexpr int BATCH    = 4;
static constexpr int SEQ      = 2048;
static constexpr int MTOK     = BATCH * SEQ;   // 8192 tokens

__device__ __forceinline__ v8f v8f_zero() {
  v8f z;
#pragma unroll
  for (int i = 0; i < 8; ++i) z[i] = 0.0f;
  return z;
}

// A-fragment (16x32 bf16). Lane L: row = L%16, half = L/16.
//   elements [0,8)  hold K = k0 + half*8 + e
//   elements [8,16) hold K = k0 + 16 + half*8 + (e-8)
// -> two contiguous 16-byte loads from the row.
__device__ __forceinline__ v16bf load_a_frag(const bf16* row, int k0, int half) {
  const v8bf a0 = *(const v8bf*)(row + k0 + half * 8);
  const v8bf a1 = *(const v8bf*)(row + k0 + 16 + half * 8);
  v16bf a;
#pragma unroll
  for (int i = 0; i < 8; ++i) { a[i] = a0[i]; a[i + 8] = a1[i]; }
  return a;
}

// B-fragment (32x16 bf16) sourced from W stored as [N][K] row-major
// (i.e. B = W^T). Lane L: column n = L%16, half = L/16.
//   elements [0,8)  hold K = k0 + half*8 + e
//   elements [8,16) hold K = k0 + 16 + half*8 + (e-8)
// -> two contiguous 16-byte reads of row n of W.
__device__ __forceinline__ v16bf load_b_frag(const bf16* row, int k0, int half) {
  const v8bf b0 = *(const v8bf*)(row + k0 + half * 8);
  const v8bf b1 = *(const v8bf*)(row + k0 + 16 + half * 8);
  v16bf b;
#pragma unroll
  for (int i = 0; i < 8; ++i) { b[i] = b0[i]; b[i + 8] = b1[i]; }
  return b;
}

// ---------------------------------------------------------------------------
// fp32 -> bf16 conversion (grid-stride)
// ---------------------------------------------------------------------------
__global__ void f32_to_bf16_kernel(const float* __restrict__ in,
                                   bf16* __restrict__ out, size_t n) {
  size_t i = (size_t)blockIdx.x * blockDim.x + threadIdx.x;
  const size_t stride = (size_t)gridDim.x * blockDim.x;
  for (; i < n; i += stride) out[i] = (bf16)in[i];
}

// ---------------------------------------------------------------------------
// Generic GEMM: C[M,N] = A[M,K](bf16) * W[N,K]^T(bf16) + bias, with epilogues.
// Block = 256 threads = 8 waves, 2(m) x 4(n); each wave owns 32 x (16*NSUB).
// NSUB=4: block tile 64x256 (8 wmma : 12 b128 per wave k-step)
// NSUB=2: block tile 64x128 (for narrow N)
// mode 0: fp32 row-major out (optional residual add)
// mode 1: bf16 row-major out (optional relu)
// mode 2: bf16 head-split   [B,H,L,64]   (for Q, K)
// mode 3: bf16 head-split-T [B,H,64,L]   (for V^T)
// ---------------------------------------------------------------------------
template <int NSUB>
__global__ __launch_bounds__(256) void gemm_wmma_bf16(
    const bf16* __restrict__ A, int lda,
    const bf16* __restrict__ W, int ldw,
    const float* __restrict__ bias,
    int M, int N, int K,
    int mode, int do_relu,
    float* __restrict__ outF, int ldoF,
    const float* __restrict__ resid, int ldr,
    bf16* __restrict__ outB, int ldoB)
{
  const int tid  = threadIdx.x;
  const int wave = tid >> 5;
  const int lane = tid & 31;
  const int half = lane >> 4;
  const int ln16 = lane & 15;
  const int mw = wave & 1;
  const int nw = wave >> 1;
  const int row0 = blockIdx.y * 64 + mw * 32;
  const int col0 = blockIdx.x * (64 * NSUB) + nw * (16 * NSUB);
  if (row0 >= M || col0 >= N) return;   // no barriers below: early-out is safe

  const bf16* arow0 = A + (size_t)(row0 + ln16) * lda;
  const bf16* arow1 = A + (size_t)(row0 + 16 + ln16) * lda;
  const bf16* wrow[NSUB];
#pragma unroll
  for (int ns = 0; ns < NSUB; ++ns)
    wrow[ns] = W + (size_t)(col0 + ns * 16 + ln16) * ldw;

  v8f c0[NSUB], c1[NSUB];
#pragma unroll
  for (int ns = 0; ns < NSUB; ++ns) { c0[ns] = v8f_zero(); c1[ns] = v8f_zero(); }

  for (int k0 = 0; k0 < K; k0 += 32) {
    const v16bf a0 = load_a_frag(arow0, k0, half);
    const v16bf a1 = load_a_frag(arow1, k0, half);
#pragma unroll
    for (int ns = 0; ns < NSUB; ++ns) {
      const v16bf b = load_b_frag(wrow[ns], k0, half);
      c0[ns] = WMMA_BF16(a0, b, c0[ns]);
      c1[ns] = WMMA_BF16(a1, b, c1[ns]);
    }
  }

#pragma unroll
  for (int ns = 0; ns < NSUB; ++ns) {
    const int col = col0 + ns * 16 + ln16;
    const float bv = bias ? bias[col] : 0.0f;
#pragma unroll
    for (int ms = 0; ms < 2; ++ms) {
#pragma unroll
      for (int r = 0; r < 8; ++r) {
        const int row = row0 + ms * 16 + half * 8 + r;   // C layout: M = half*8 + r
        float v = (ms == 0 ? c0[ns][r] : c1[ns][r]) + bv;
        if (do_relu) v = fmaxf(v, 0.0f);
        if (resid) v += resid[(size_t)row * ldr + col];
        if (mode == 0) {
          outF[(size_t)row * ldoF + col] = v;
        } else if (mode == 1) {
          outB[(size_t)row * ldoB + col] = (bf16)v;
        } else {
          const int bidx = row >> 11;        // SEQ = 2048
          const int l    = row & 2047;
          const int hh   = col >> 6;         // HEAD_DIM = 64
          const int d    = col & 63;
          size_t idx;
          if (mode == 2) idx = (((size_t)(bidx * N_HEADS + hh) * SEQ) + l) * HEAD_DIM + d;
          else           idx = (((size_t)(bidx * N_HEADS + hh) * HEAD_DIM) + d) * SEQ + l;
          outB[idx] = (bf16)v;
        }
      }
    }
  }
}

// ---------------------------------------------------------------------------
// Flash attention: grid = (Lq/64, B*H), block = 128 (4 waves x 16 q-rows).
// Q: [B,H,Lq,64] bf16, K: [B,H,Lk,64] bf16, Vt: [B,H,64,Lk] bf16.
// 64-wide KV tiles: 16 WMMAs per online-softmax pass; next K/V tile is
// prefetched (global_prefetch_b8) since the softmax dependency chain blocks
// natural load/compute overlap. P converts C-layout -> A-layout via a
// per-wave 2KB LDS tile. O written bf16 token-major [MTOK, 512].
// ---------------------------------------------------------------------------
__global__ __launch_bounds__(128) void flash_attn_wmma(
    const bf16* __restrict__ Q, const bf16* __restrict__ Kmat,
    const bf16* __restrict__ Vt, bf16* __restrict__ O,
    int Lq, int Lk)
{
  __shared__ bf16 lds_p[4][16][64];
  const int tid  = threadIdx.x;
  const int wave = tid >> 5;
  const int lane = tid & 31;
  const int half = lane >> 4;
  const int ln16 = lane & 15;
  const int bh = blockIdx.y;                 // b*N_HEADS + h
  const int q0 = blockIdx.x * 64 + wave * 16;

  const bf16* qrow  = Q + ((size_t)bh * Lq + q0 + ln16) * HEAD_DIM;
  const bf16* kbase = Kmat + (size_t)bh * Lk * HEAD_DIM;
  const bf16* vbase = Vt + (size_t)bh * HEAD_DIM * Lk;

  const v16bf qf0 = load_a_frag(qrow, 0, half);
  const v16bf qf1 = load_a_frag(qrow, 32, half);

  v8f o[4];
#pragma unroll
  for (int nt = 0; nt < 4; ++nt) o[nt] = v8f_zero();
  float mi[8], li[8];
#pragma unroll
  for (int r = 0; r < 8; ++r) { mi[r] = -1.0e30f; li[r] = 0.0f; }
  const float scale = 0.125f;   // 1/sqrt(64)

  for (int kv = 0; kv < Lk; kv += 64) {
    // Prefetch the next 64-wide K/V tile while this iteration's softmax runs.
    const int kvn = kv + 64;
    if (kvn < Lk) {
#pragma unroll
      for (int nt = 0; nt < 4; ++nt) {
        __builtin_prefetch(kbase + (size_t)(kvn + nt * 16 + ln16) * HEAD_DIM, 0, 3);
        __builtin_prefetch(vbase + (size_t)(nt * 16 + ln16) * Lk + kvn, 0, 3);
      }
    }

    // Scores S[16 x 64] = Q (16x64) x K_tile^T, 8 WMMAs.
    v8f s[4];
#pragma unroll
    for (int nt = 0; nt < 4; ++nt) s[nt] = v8f_zero();
#pragma unroll
    for (int nt = 0; nt < 4; ++nt) {
      const bf16* kr = kbase + (size_t)(kv + nt * 16 + ln16) * HEAD_DIM;
      s[nt] = WMMA_BF16(qf0, load_b_frag(kr, 0, half), s[nt]);
      s[nt] = WMMA_BF16(qf1, load_b_frag(kr, 32, half), s[nt]);
    }

    // Online softmax (rows half*8+r live across 16 lanes of each half).
    float alpha[8];
#pragma unroll
    for (int r = 0; r < 8; ++r) {
      float mx = fmaxf(fmaxf(s[0][r], s[1][r]), fmaxf(s[2][r], s[3][r])) * scale;
      mx = fmaxf(mx, __shfl_xor(mx, 1));   // xor < 16 stays within the 16-lane half
      mx = fmaxf(mx, __shfl_xor(mx, 2));
      mx = fmaxf(mx, __shfl_xor(mx, 4));
      mx = fmaxf(mx, __shfl_xor(mx, 8));
      const float mnew = fmaxf(mi[r], mx);
      alpha[r] = __expf(mi[r] - mnew);
      mi[r] = mnew;
      float ps = 0.0f;
#pragma unroll
      for (int nt = 0; nt < 4; ++nt) {
        const float p = __expf(s[nt][r] * scale - mnew);
        lds_p[wave][half * 8 + r][nt * 16 + ln16] = (bf16)p;
        ps += p;
      }
      ps += __shfl_xor(ps, 1);
      ps += __shfl_xor(ps, 2);
      ps += __shfl_xor(ps, 4);
      ps += __shfl_xor(ps, 8);
      li[r] = li[r] * alpha[r] + ps;
    }
#pragma unroll
    for (int nt = 0; nt < 4; ++nt)
#pragma unroll
      for (int r = 0; r < 8; ++r) o[nt][r] *= alpha[r];

    // per-wave LDS region; DS ops from one wave are in order, keep the
    // compiler from reordering across the write->read.
    __builtin_amdgcn_wave_barrier();
    const v16bf pa0 = load_a_frag(&lds_p[wave][ln16][0], 0, half);
    const v16bf pa1 = load_a_frag(&lds_p[wave][ln16][0], 32, half);
    __builtin_amdgcn_wave_barrier();

    // O += P(16x64) x V_tile(64x64), 8 WMMAs against V^T rows.
#pragma unroll
    for (int nt = 0; nt < 4; ++nt) {
      const bf16* vrow = vbase + (size_t)(nt * 16 + ln16) * Lk;
      o[nt] = WMMA_BF16(pa0, load_b_frag(vrow, kv, half), o[nt]);
      o[nt] = WMMA_BF16(pa1, load_b_frag(vrow, kv + 32, half), o[nt]);
    }
  }

  const int b = bh >> 3;   // N_HEADS = 8
  const int h = bh & 7;
#pragma unroll
  for (int r = 0; r < 8; ++r) {
    const float inv = 1.0f / li[r];
    const size_t mtok = (size_t)b * Lq + q0 + half * 8 + r;
#pragma unroll
    for (int nt = 0; nt < 4; ++nt) {
      const int col = h * HEAD_DIM + nt * 16 + ln16;
      O[mtok * D_MODEL + col] = (bf16)(o[nt][r] * inv);
    }
  }
}

// ---------------------------------------------------------------------------
// LayerNorm over D_MODEL=512; one block (128 threads, 4 floats each) per row.
// Writes fp32 out, and optionally a bf16 copy for the next GEMM stage.
// ---------------------------------------------------------------------------
__global__ __launch_bounds__(128) void layernorm_kernel(
    const float* __restrict__ in, const float* __restrict__ gamma,
    const float* __restrict__ beta, float* __restrict__ outF,
    bf16* __restrict__ outB)
{
  __shared__ float ls[4], ls2[4];
  const int row = blockIdx.x;
  const int t = threadIdx.x;
  const float4 v = ((const float4*)(in + (size_t)row * D_MODEL))[t];
  float s  = v.x + v.y + v.z + v.w;
  float s2 = v.x * v.x + v.y * v.y + v.z * v.z + v.w * v.w;
#pragma unroll
  for (int m = 1; m < 32; m <<= 1) { s += __shfl_xor(s, m); s2 += __shfl_xor(s2, m); }
  const int w = t >> 5;
  if ((t & 31) == 0) { ls[w] = s; ls2[w] = s2; }
  __syncthreads();
  s  = ls[0] + ls[1] + ls[2] + ls[3];
  s2 = ls2[0] + ls2[1] + ls2[2] + ls2[3];
  const float mu  = s * (1.0f / D_MODEL);
  const float var = s2 * (1.0f / D_MODEL) - mu * mu;
  const float inv = rsqrtf(var + 1e-5f);
  const float4 g4 = ((const float4*)gamma)[t];
  const float4 b4 = ((const float4*)beta)[t];
  float4 ov;
  ov.x = (v.x - mu) * inv * g4.x + b4.x;
  ov.y = (v.y - mu) * inv * g4.y + b4.y;
  ov.z = (v.z - mu) * inv * g4.z + b4.z;
  ov.w = (v.w - mu) * inv * g4.w + b4.w;
  ((float4*)(outF + (size_t)row * D_MODEL))[t] = ov;
  if (outB) {
    bf16* ob = outB + (size_t)row * D_MODEL + t * 4;
    ob[0] = (bf16)ov.x; ob[1] = (bf16)ov.y; ob[2] = (bf16)ov.z; ob[3] = (bf16)ov.w;
  }
}

// ---------------------------------------------------------------------------
// Host: full decoder layer pipeline.
// ---------------------------------------------------------------------------
extern "C" void kernel_launch(void* const* d_in, const int* in_sizes, int n_in,
                              void* d_out, int out_size, void* d_ws, size_t ws_size,
                              hipStream_t stream)
{
  (void)in_sizes; (void)n_in; (void)out_size; (void)ws_size;

  const float* x_in  = (const float*)d_in[0];
  const float* cr_in = (const float*)d_in[1];
  // order: self {wq,bq,wk,bk,wv,bv,wo,bo}, cross {...}, ln1..3 {g,b}, ff_w1,ff_b1,ff_w2,ff_b2
  const float* wA[8]; const float* bA[8];
  for (int i = 0; i < 8; ++i) {
    wA[i] = (const float*)d_in[2 + 2 * i];
    bA[i] = (const float*)d_in[3 + 2 * i];
  }
  const float* lng[3] = {(const float*)d_in[18], (const float*)d_in[20], (const float*)d_in[22]};
  const float* lnb[3] = {(const float*)d_in[19], (const float*)d_in[21], (const float*)d_in[23]};
  const float* ffw1 = (const float*)d_in[24];
  const float* ffb1 = (const float*)d_in[25];
  const float* ffw2 = (const float*)d_in[26];
  const float* ffb2 = (const float*)d_in[27];
  float* out = (float*)d_out;

  // bump allocator over workspace (all sub-buffers 256B aligned)
  char* base = (char*)d_ws;
  size_t off = 0;
  auto alloc = [&](size_t bytes) -> void* {
    void* p = base + off;
    off = (off + bytes + 255) & ~(size_t)255;
    return p;
  };
  bf16*  xb   = (bf16*)alloc((size_t)MTOK * D_MODEL * 2);
  bf16*  crb  = (bf16*)alloc((size_t)MTOK * D_MODEL * 2);
  bf16*  wAb  = (bf16*)alloc((size_t)8 * D_MODEL * D_MODEL * 2);
  bf16*  w1b  = (bf16*)alloc((size_t)N_HEADS * FF_DIM * HEAD_DIM * 2);
  bf16*  w2b  = (bf16*)alloc((size_t)N_HEADS * HEAD_DIM * FF_DIM * 2);
  bf16*  qb   = (bf16*)alloc((size_t)MTOK * D_MODEL * 2);
  bf16*  kb   = (bf16*)alloc((size_t)MTOK * D_MODEL * 2);
  bf16*  vtb  = (bf16*)alloc((size_t)MTOK * D_MODEL * 2);
  bf16*  aob  = (bf16*)alloc((size_t)MTOK * D_MODEL * 2);
  float* tbuf = (float*)alloc((size_t)MTOK * D_MODEL * 4);
  float* x1   = (float*)alloc((size_t)MTOK * D_MODEL * 4);
  bf16*  x1b  = (bf16*)alloc((size_t)MTOK * D_MODEL * 2);
  float* x2   = (float*)alloc((size_t)MTOK * D_MODEL * 4);
  bf16*  x2b  = (bf16*)alloc((size_t)MTOK * D_MODEL * 2);
  bf16*  h1b  = (bf16*)alloc((size_t)MTOK * FF_DIM * 2);

  auto conv = [&](const float* src, bf16* dst, size_t n) {
    f32_to_bf16_kernel<<<1024, 256, 0, stream>>>(src, dst, n);
  };
  auto gemm = [&](const bf16* A, int lda, const bf16* W, int ldw, const float* bias,
                  int M, int N, int K, int mode, int relu,
                  float* oF, int ldoF, const float* res, int ldr, bf16* oB, int ldoB) {
    if (N >= 256) {
      dim3 g((N + 255) / 256, M / 64);
      gemm_wmma_bf16<4><<<g, 256, 0, stream>>>(A, lda, W, ldw, bias, M, N, K, mode, relu,
                                               oF, ldoF, res, ldr, oB, ldoB);
    } else {
      dim3 g((N + 127) / 128, M / 64);
      gemm_wmma_bf16<2><<<g, 256, 0, stream>>>(A, lda, W, ldw, bias, M, N, K, mode, relu,
                                               oF, ldoF, res, ldr, oB, ldoB);
    }
  };
  auto lnorm = [&](const float* in, const float* g, const float* b, float* oF, bf16* oB) {
    layernorm_kernel<<<MTOK, 128, 0, stream>>>(in, g, b, oF, oB);
  };
  auto wptr = [&](int i) { return wAb + (size_t)i * D_MODEL * D_MODEL; };

  // ---- precision downconvert (activations + weights) ----
  conv(x_in,  xb,  (size_t)MTOK * D_MODEL);
  conv(cr_in, crb, (size_t)MTOK * D_MODEL);
  for (int i = 0; i < 8; ++i) conv(wA[i], wptr(i), (size_t)D_MODEL * D_MODEL);
  conv(ffw1, w1b, (size_t)N_HEADS * FF_DIM * HEAD_DIM);
  conv(ffw2, w2b, (size_t)N_HEADS * HEAD_DIM * FF_DIM);

  const dim3 fgrid(SEQ / 64, BATCH * N_HEADS);

  // ---- self attention ----
  gemm(xb, D_MODEL, wptr(0), D_MODEL, bA[0], MTOK, D_MODEL, D_MODEL, 2, 0,
       nullptr, 0, nullptr, 0, qb, 0);
  gemm(xb, D_MODEL, wptr(1), D_MODEL, bA[1], MTOK, D_MODEL, D_MODEL, 2, 0,
       nullptr, 0, nullptr, 0, kb, 0);
  gemm(xb, D_MODEL, wptr(2), D_MODEL, bA[2], MTOK, D_MODEL, D_MODEL, 3, 0,
       nullptr, 0, nullptr, 0, vtb, 0);
  flash_attn_wmma<<<fgrid, 128, 0, stream>>>(qb, kb, vtb, aob, SEQ, SEQ);
  gemm(aob, D_MODEL, wptr(3), D_MODEL, bA[3], MTOK, D_MODEL, D_MODEL, 0, 0,
       tbuf, D_MODEL, x_in, D_MODEL, nullptr, 0);          // + residual x
  lnorm(tbuf, lng[0], lnb[0], x1, x1b);

  // ---- cross attention ----
  gemm(x1b, D_MODEL, wptr(4), D_MODEL, bA[4], MTOK, D_MODEL, D_MODEL, 2, 0,
       nullptr, 0, nullptr, 0, qb, 0);
  gemm(crb, D_MODEL, wptr(5), D_MODEL, bA[5], MTOK, D_MODEL, D_MODEL, 2, 0,
       nullptr, 0, nullptr, 0, kb, 0);
  gemm(crb, D_MODEL, wptr(6), D_MODEL, bA[6], MTOK, D_MODEL, D_MODEL, 3, 0,
       nullptr, 0, nullptr, 0, vtb, 0);
  flash_attn_wmma<<<fgrid, 128, 0, stream>>>(qb, kb, vtb, aob, SEQ, SEQ);
  gemm(aob, D_MODEL, wptr(7), D_MODEL, bA[7], MTOK, D_MODEL, D_MODEL, 0, 0,
       tbuf, D_MODEL, x1, D_MODEL, nullptr, 0);            // + residual x1
  lnorm(tbuf, lng[1], lnb[1], x2, x2b);

  // ---- per-head FFN: h = relu(x2_h @ w1[h]^T + b1), y = h @ w2[h]^T + b2 + x2 ----
  for (int h = 0; h < N_HEADS; ++h) {
    gemm(x2b + h * HEAD_DIM, D_MODEL,
         w1b + (size_t)h * FF_DIM * HEAD_DIM, HEAD_DIM,
         ffb1 + (size_t)h * FF_DIM,
         MTOK, FF_DIM, HEAD_DIM, 1, 1,
         nullptr, 0, nullptr, 0, h1b, FF_DIM);
    gemm(h1b, FF_DIM,
         w2b + (size_t)h * HEAD_DIM * FF_DIM, FF_DIM,
         ffb2 + (size_t)h * HEAD_DIM,
         MTOK, HEAD_DIM, FF_DIM, 0, 0,
         tbuf + h * HEAD_DIM, D_MODEL, x2 + h * HEAD_DIM, D_MODEL, nullptr, 0);
  }
  lnorm(tbuf, lng[2], lnb[2], out, nullptr);
}
